// MultiHeadSelfAttention_84224308674728
// MI455X (gfx1250) — compile-verified, matrix operand order checked by reading, not run on hardware
//
#include <hip/hip_runtime.h>
#include <hip/hip_bf16.h>

// Problem constants (match reference)
static constexpr int B_ = 2, T_ = 2048, E_ = 1024, H_ = 16, M_ = 16, D_ = 64;
static constexpr int BT = B_ * T_;          // 4096 rows
static constexpr int N_QKV = 3 * E_;        // 3072
static constexpr int BH = B_ * H_;          // 32

typedef __attribute__((ext_vector_type(16))) __bf16 v16bf;
typedef __attribute__((ext_vector_type(8)))  float  v8f;

__device__ __forceinline__ unsigned short f2bf_bits(float x) {
  unsigned int u = __builtin_bit_cast(unsigned int, x);
  u += 0x7FFFu + ((u >> 16) & 1u);          // round-to-nearest-even
  return (unsigned short)(u >> 16);
}

__device__ __forceinline__ v8f wmma_bf16(v16bf a, v16bf b, v8f c) {
  return __builtin_amdgcn_wmma_f32_16x16x32_bf16(
      /*neg_a=*/false, a, /*neg_b=*/false, b,
      /*c_mod=*/(short)0, c, /*reuse_a=*/false, /*reuse_b=*/false);
}

// Load a 16x32 bf16 operand fragment (A layout; B uses the symmetric layout
// with "row" = output column). src points at (tile_row0, k0); elements are
// row-major with leading dimension `ld` (bf16 units, even => dword aligned).
// Lane L: row = L%16; K-base kb = 8*(L/16); VGPR v in 0..3 holds K=kb+2v,+1;
// VGPR v in 4..7 holds K=16+kb+2(v-4),+1.  (CDNA5 ISA 7.12.2)
template <typename PT>
__device__ __forceinline__ v16bf load_frag(const PT* src, int ld) {
  int lane = threadIdx.x & 31;
  const PT* rowp = src + (lane & 15) * ld;
  int kb = (lane >> 4) << 3;
  v16bf f;
  unsigned int* w = reinterpret_cast<unsigned int*>(&f);
#pragma unroll
  for (int v = 0; v < 4; ++v)
    w[v] = *reinterpret_cast<const unsigned int*>(rowp + kb + 2 * v);
#pragma unroll
  for (int v = 0; v < 4; ++v)
    w[4 + v] = *reinterpret_cast<const unsigned int*>(rowp + 16 + kb + 2 * v);
  return f;
}

// ---------------------------------------------------------------- converts
// Vectorized: 4 floats in (b128), 4 bf16 out (b64). All sizes % 1024 == 0.
__global__ void cvt_f32_bf16(const float* __restrict__ src,
                             unsigned short* __restrict__ dst, int n4) {
  int i = blockIdx.x * blockDim.x + threadIdx.x;
  if (i < n4) {
    float4 f = reinterpret_cast<const float4*>(src)[i];
    ushort4 o;
    o.x = f2bf_bits(f.x); o.y = f2bf_bits(f.y);
    o.z = f2bf_bits(f.z); o.w = f2bf_bits(f.w);
    reinterpret_cast<ushort4*>(dst)[i] = o;
  }
}

// ------------------------------------------------- GEMM1: qkv = x @ Wqkv^T
// C[m,n] = sum_k X[m,k] * Wqkv[n,k]; M=4096, N=3072, K=1024.
// 2x2 register blocking: each wave computes a 32x32 output block.
__device__ __forceinline__ void store_qkv_tile(
    v8f acc, int m0, int n0, const float* __restrict__ bqkv,
    unsigned short* __restrict__ Qb, unsigned short* __restrict__ Kb,
    unsigned short* __restrict__ Vt) {
  int lane = threadIdx.x & 31;
  int nloc = lane & 15, half = lane >> 4;
  int n = n0 + nloc;
  float bias = bqkv[n];
  int which = n >> 10, h = (n & 1023) >> 6, d = n & 63;
#pragma unroll
  for (int v = 0; v < 8; ++v) {
    int m = m0 + v + half * 8;
    int b_ = m >> 11, t = m & (T_ - 1);
    int bh = b_ * H_ + h;
    float c = acc[v] + bias;
    if (which == 0)      Qb[((size_t)bh * T_ + t) * D_ + d] = f2bf_bits(c * 0.125f);
    else if (which == 1) Kb[((size_t)bh * T_ + t) * D_ + d] = f2bf_bits(c);
    else                 Vt[((size_t)bh * D_ + d) * T_ + t] = f2bf_bits(c);
  }
}

__global__ __launch_bounds__(128) void gemm_qkv_k(
    const unsigned short* __restrict__ Xb, const unsigned short* __restrict__ Wb,
    const float* __restrict__ bqkv, unsigned short* __restrict__ Qb,
    unsigned short* __restrict__ Kb, unsigned short* __restrict__ Vt) {
  int wave = threadIdx.x >> 5;
  int tile = blockIdx.x * 4 + wave;          // 128*96 = 12288 wave-tiles
  int tm = tile / (N_QKV / 32);              // 0..127 (32-row block)
  int tn = tile % (N_QKV / 32);              // 0..95  (32-col block)
  const unsigned short* a0 = Xb + (size_t)tm * 32 * E_;
  const unsigned short* a1 = a0 + (size_t)16 * E_;
  const unsigned short* b0 = Wb + (size_t)tn * 32 * E_;
  const unsigned short* b1 = b0 + (size_t)16 * E_;
  v8f acc00 = {}, acc01 = {}, acc10 = {}, acc11 = {};
#pragma unroll 2
  for (int k = 0; k < E_; k += 32) {
    v16bf A0 = load_frag(a0 + k, E_);
    v16bf A1 = load_frag(a1 + k, E_);
    v16bf B0 = load_frag(b0 + k, E_);
    v16bf B1 = load_frag(b1 + k, E_);
    acc00 = wmma_bf16(A0, B0, acc00);
    acc01 = wmma_bf16(A0, B1, acc01);
    acc10 = wmma_bf16(A1, B0, acc10);
    acc11 = wmma_bf16(A1, B1, acc11);
  }
  store_qkv_tile(acc00, tm * 32,      tn * 32,      bqkv, Qb, Kb, Vt);
  store_qkv_tile(acc01, tm * 32,      tn * 32 + 16, bqkv, Qb, Kb, Vt);
  store_qkv_tile(acc10, tm * 32 + 16, tn * 32,      bqkv, Qb, Kb, Vt);
  store_qkv_tile(acc11, tm * 32 + 16, tn * 32 + 16, bqkv, Qb, Kb, Vt);
}

// ----------------------------------- fused scores + bias + softmax + P@V
// One block per (b,h, 16-row tile). 4 waves. Dynamic LDS:
//   sc  : 16x2048 f32 scores/probs (128 KB)
//   pb  : 16x2048 bf16 normalized P (64 KB)
//   red : 16x8 partials, rowmax[16], rowsum[16], btab[33]
__global__ __launch_bounds__(128) void attn_k(
    const unsigned short* __restrict__ Qb, const unsigned short* __restrict__ Kb,
    const unsigned short* __restrict__ Vt, const unsigned char* __restrict__ maskp,
    const float* __restrict__ rel_table, float* __restrict__ attn,
    unsigned short* __restrict__ OH) {
  extern __shared__ char smem[];
  float* sc = (float*)smem;                                    // 16*2048 f32
  unsigned short* pb = (unsigned short*)(smem + 16 * T_ * 4);  // 16*2048 bf16
  float* red = (float*)(smem + 16 * T_ * 4 + 16 * T_ * 2);     // 128
  float* rowmax = red + 128;                                   // 16
  float* rowsum = rowmax + 16;                                 // 16
  float* btab = rowsum + 16;                                   // 33

  int blk = blockIdx.x;
  int mt = blk & (T_ / 16 - 1);
  int bh = blk >> 7;
  int h = bh & (H_ - 1), b = bh >> 4;
  int t0 = mt << 4;
  int tid = threadIdx.x, wave = tid >> 5, lane = tid & 31;
  int nloc = lane & 15, half = lane >> 4;

  if (tid < 2 * M_ + 1) btab[tid] = rel_table[tid * H_ + h];
  __syncthreads();

  // Q fragments for this 16-row tile (K = D = 64 -> two k-steps)
  const unsigned short* qbase = Qb + ((size_t)bh * T_ + t0) * D_;
  v16bf qa0 = load_frag(qbase, D_);
  v16bf qa1 = load_frag(qbase + 32, D_);

  const unsigned char* mrow = maskp + (size_t)b * T_;
  bool mi8[8];
#pragma unroll
  for (int v = 0; v < 8; ++v) mi8[v] = mrow[t0 + v + half * 8] != 0;

  // Phase 1: scores tile row (16 x 2048), 32 column-tiles per wave
  for (int jt = wave; jt < T_ / 16; jt += 4) {
    const unsigned short* kbase = Kb + ((size_t)bh * T_ + jt * 16) * D_;
    v16bf kb0 = load_frag(kbase, D_);
    v16bf kb1 = load_frag(kbase + 32, D_);
    v8f acc = {};
    acc = wmma_bf16(qa0, kb0, acc);
    acc = wmma_bf16(qa1, kb1, acc);
    int j = jt * 16 + nloc;
    bool mj = mrow[j] != 0;
#pragma unroll
    for (int v = 0; v < 8; ++v) {
      int i = v + half * 8;
      int ig = t0 + i;
      int dlt = ig - j;
      dlt = dlt > M_ ? M_ : (dlt < -M_ ? -M_ : dlt);
      float s = acc[v] + btab[dlt + M_];
      sc[i * T_ + j] = (mj && mi8[v]) ? s : -1e9f;
    }
  }
  __syncthreads();

  // Phase 2: softmax. 8 threads per row, 256 strided elements each.
  int row = tid >> 3, sub = tid & 7;
  float mx = -3.4e38f;
  for (int c = 0; c < T_ / 8; ++c) mx = fmaxf(mx, sc[row * T_ + sub + c * 8]);
  red[row * 8 + sub] = mx;
  __syncthreads();
  if (sub == 0) {
    float m2 = red[row * 8];
#pragma unroll
    for (int q = 1; q < 8; ++q) m2 = fmaxf(m2, red[row * 8 + q]);
    rowmax[row] = m2;
  }
  __syncthreads();
  float rmax = rowmax[row];
  float sum = 0.f;
  for (int c = 0; c < T_ / 8; ++c) {
    int idx = row * T_ + sub + c * 8;
    float e = __expf(sc[idx] - rmax);
    sc[idx] = e;
    sum += e;
  }
  red[row * 8 + sub] = sum;
  __syncthreads();
  if (sub == 0) {
    float s2 = 0.f;
#pragma unroll
    for (int q = 0; q < 8; ++q) s2 += red[row * 8 + q];
    rowsum[row] = s2;
  }
  __syncthreads();
  float inv = 1.0f / rowsum[row];
  size_t abase = ((size_t)bh * T_ + (t0 + row)) * T_;
  for (int c = 0; c < T_ / 8; ++c) {
    int jj = sub + c * 8;
    float p = sc[row * T_ + jj] * inv;
    attn[abase + jj] = p;
    pb[row * T_ + jj] = f2bf_bits(p);
  }
  __syncthreads();

  // Phase 3: out = P @ V. Wave w owns d-tile w (16 of the 64 d's).
  int dt = wave;
  const unsigned short* vbase = Vt + ((size_t)bh * D_ + dt * 16) * T_;
  v8f oacc = {};
#pragma unroll 2
  for (int jt = 0; jt < T_ / 32; ++jt) {
    v16bf pa = load_frag(pb + jt * 32, T_);      // LDS, row-major
    v16bf vb = load_frag(vbase + jt * 32, T_);   // Vt: row = d, k-contiguous
    oacc = wmma_bf16(pa, vb, oacc);
  }
  int e = h * D_ + dt * 16 + nloc;
#pragma unroll
  for (int v = 0; v < 8; ++v) {
    int t = t0 + v + half * 8;
    OH[((size_t)b * T_ + t) * E_ + e] = f2bf_bits(oacc[v]);
  }
}

// --------------------------------------- GEMM3: out = OH @ Wout^T + b_out
__device__ __forceinline__ void store_out_tile(
    v8f acc, int m0, int n0, const float* __restrict__ bout,
    float* __restrict__ out) {
  int lane = threadIdx.x & 31;
  int nloc = lane & 15, half = lane >> 4;
  int n = n0 + nloc;
  float bias = bout[n];
#pragma unroll
  for (int v = 0; v < 8; ++v) {
    int m = m0 + v + half * 8;
    out[(size_t)m * E_ + n] = acc[v] + bias;
  }
}

__global__ __launch_bounds__(128) void gemm_out_k(
    const unsigned short* __restrict__ OH, const unsigned short* __restrict__ Wob,
    const float* __restrict__ bout, float* __restrict__ out) {
  int wave = threadIdx.x >> 5;
  int tile = blockIdx.x * 4 + wave;          // 128*32 = 4096 wave-tiles
  int tm = tile >> 5, tn = tile & 31;        // 32-row / 32-col blocks
  const unsigned short* a0 = OH + (size_t)tm * 32 * E_;
  const unsigned short* a1 = a0 + (size_t)16 * E_;
  const unsigned short* b0 = Wob + (size_t)tn * 32 * E_;
  const unsigned short* b1 = b0 + (size_t)16 * E_;
  v8f acc00 = {}, acc01 = {}, acc10 = {}, acc11 = {};
#pragma unroll 2
  for (int k = 0; k < E_; k += 32) {
    v16bf A0 = load_frag(a0 + k, E_);
    v16bf A1 = load_frag(a1 + k, E_);
    v16bf B0 = load_frag(b0 + k, E_);
    v16bf B1 = load_frag(b1 + k, E_);
    acc00 = wmma_bf16(A0, B0, acc00);
    acc01 = wmma_bf16(A0, B1, acc01);
    acc10 = wmma_bf16(A1, B0, acc10);
    acc11 = wmma_bf16(A1, B1, acc11);
  }
  store_out_tile(acc00, tm * 32,      tn * 32,      bout, out);
  store_out_tile(acc01, tm * 32,      tn * 32 + 16, bout, out);
  store_out_tile(acc10, tm * 32 + 16, tn * 32,      bout, out);
  store_out_tile(acc11, tm * 32 + 16, tn * 32 + 16, bout, out);
}

extern "C" void kernel_launch(void* const* d_in, const int* in_sizes, int n_in,
                              void* d_out, int out_size, void* d_ws, size_t ws_size,
                              hipStream_t stream) {
  const float* x         = (const float*)d_in[0];
  const unsigned char* m = (const unsigned char*)d_in[1];
  const float* w_qkv     = (const float*)d_in[2];
  const float* b_qkv     = (const float*)d_in[3];
  const float* w_out     = (const float*)d_in[4];
  const float* b_out     = (const float*)d_in[5];
  const float* rel_tab   = (const float*)d_in[6];

  float* out  = (float*)d_out;
  float* attn = out + (size_t)BT * E_;       // (B,H,T,T) after (B,T,E)

  // Workspace layout (bf16 = unsigned short), ~48 MB total.
  unsigned short* Xb  = (unsigned short*)d_ws;
  unsigned short* Wqb = Xb  + (size_t)BT * E_;          // 3072x1024
  unsigned short* Wob = Wqb + (size_t)N_QKV * E_;       // 1024x1024
  unsigned short* Qb  = Wob + (size_t)E_ * E_;          // (BH,T,D)
  unsigned short* Kb  = Qb  + (size_t)BH * T_ * D_;
  unsigned short* Vt  = Kb  + (size_t)BH * T_ * D_;     // (BH,D,T)
  unsigned short* OH  = Vt  + (size_t)BH * T_ * D_;     // (BT,E) bf16

  int nx4 = BT * E_ / 4, nwq4 = N_QKV * E_ / 4, nwo4 = E_ * E_ / 4;
  cvt_f32_bf16<<<(nx4  + 255) / 256, 256, 0, stream>>>(x,     Xb,  nx4);
  cvt_f32_bf16<<<(nwq4 + 255) / 256, 256, 0, stream>>>(w_qkv, Wqb, nwq4);
  cvt_f32_bf16<<<(nwo4 + 255) / 256, 256, 0, stream>>>(w_out, Wob, nwo4);

  gemm_qkv_k<<<(BT / 32) * (N_QKV / 32) / 4, 128, 0, stream>>>(
      Xb, Wqb, b_qkv, Qb, Kb, Vt);

  size_t smem = (size_t)16 * T_ * 4 + (size_t)16 * T_ * 2 + (128 + 16 + 16 + 33) * 4;
  attn_k<<<BH * (T_ / 16), 128, smem, stream>>>(Qb, Kb, Vt, m, rel_tab, attn, OH);

  gemm_out_k<<<(BT / 32) * (E_ / 32) / 4, 128, 0, stream>>>(OH, Wob, b_out, out);
}
